// CrossAttentionHead_11390253269216
// MI455X (gfx1250) — matrix operand order repaired and checked by reading, not compiled, not run on hardware
//
#include <hip/hip_runtime.h>
#include <hip/hip_bf16.h>

// CDNA5 / gfx1250: wave32, WMMA 16x16x32 bf16.
typedef __attribute__((ext_vector_type(16))) __bf16 v16bf;
typedef __attribute__((ext_vector_type(8)))  __bf16 v8bf;
typedef __attribute__((ext_vector_type(2)))  __bf16 v2bf;
typedef __attribute__((ext_vector_type(8)))  float  v8f;

#define EMBED 768
#define SEQ   1024
#define HD    64
#define NB    16

// Packed f32x2 -> bf16x2 via native v_cvt_pk_bf16_f32 (confirmed on gfx1250).
static __device__ __forceinline__ v2bf pk2bf(float a, float b) {
#if __has_builtin(__builtin_amdgcn_cvt_pk_bf16_f32)
    return __builtin_amdgcn_cvt_pk_bf16_f32(a, b);
#else
    v2bf r; r[0] = (__bf16)a; r[1] = (__bf16)b; return r;
#endif
}
static __device__ __forceinline__ __bf16 s2bf(float a) { return (__bf16)a; }

union A16 { v16bf v; v2bf p[8]; };

// ---------------------------------------------------------------------------
// Prep: convert Wq|Wk|Wv (3 x 64x768 f32) to bf16 once.
// ---------------------------------------------------------------------------
__global__ void __launch_bounds__(256)
xattn_prep_w(const float* __restrict__ Wq, const float* __restrict__ Wk,
             const float* __restrict__ Wv, __bf16* __restrict__ Wb) {
    const int per   = HD * EMBED;                       // 49152
    const int t     = blockIdx.x * 256 + threadIdx.x;   // pair index
    const int npair = 3 * per / 2;                      // 73728
    if (t >= npair) return;
    const int m = t * 2;
    const float* src = (m < per) ? Wq : (m < 2 * per ? Wk : Wv);
    const int off = m % per;
    *(v2bf*)(Wb + m) = pk2bf(src[off], src[off + 1]);
}

// ---------------------------------------------------------------------------
// Projection: P[16x64] = X[16x768] @ W^T + b, per wave. gridDim.y: 0=Q,1=K,2=V.
// Q,K stored bf16 row-major [B][SEQ][64]; V stored bf16 transposed [B][64][SEQ].
// ---------------------------------------------------------------------------
__global__ void __launch_bounds__(128)
xattn_proj_kernel(const float* __restrict__ dec, const float* __restrict__ enc,
                  const __bf16* __restrict__ Wb,
                  const float* __restrict__ bq, const float* __restrict__ bk,
                  const float* __restrict__ bv,
                  __bf16* __restrict__ Qb, __bf16* __restrict__ Kb,
                  __bf16* __restrict__ Vt) {
    const int lane  = threadIdx.x & 31;
    const int wid   = threadIdx.x >> 5;
    const int tile  = blockIdx.x * 4 + wid;          // 0..1023 row tiles
    const int b     = tile >> 6;
    const int r0    = (tile & 63) << 4;              // seq row base
    const int proj  = blockIdx.y;                    // 0=Q 1=K 2=V

    const float*  X    = (proj == 0 ? dec : enc) + ((size_t)b * SEQ + r0) * EMBED;
    const __bf16* W    = Wb + (size_t)proj * HD * EMBED;
    const float*  bias = (proj == 0) ? bq : (proj == 1 ? bk : bv);

    const int n     = lane & 15;          // N col within tile / A row
    const int hg    = lane >> 4;          // half-group
    const int abase = hg ? 16 : 0;

    const float* xrow = X + (size_t)n * EMBED;

    v8f acc[4];
    #pragma unroll
    for (int nt = 0; nt < 4; ++nt) acc[nt] = v8f{0,0,0,0,0,0,0,0};

    for (int kk = 0; kk < EMBED; kk += 32) {
        // A fragment (16x32 bf16): row = lane&15; K = 16*hg + i, contiguous like B.
        A16 a;
        #pragma unroll
        for (int i = 0; i < 4; ++i) {
            a.p[i]     = pk2bf(xrow[kk + abase + 2*i],      xrow[kk + abase + 2*i + 1]);
            a.p[i + 4] = pk2bf(xrow[kk + abase + 8 + 2*i], xrow[kk + abase + 8 + 2*i + 1]);
        }
        #pragma unroll
        for (int nt = 0; nt < 4; ++nt) {
            // B fragment (32x16): col = lane&15, K = 16*hg + i, contiguous bf16.
            v16bf bm = *(const v16bf*)(W + (size_t)(nt * 16 + n) * EMBED + kk + hg * 16);
            acc[nt] = __builtin_amdgcn_wmma_f32_16x16x32_bf16(
                false, a.v, false, bm, (short)0, acc[nt], false, false);
        }
    }

    // Add bias, convert to bf16, store. C layout: row = r + 8*hg, col = lane&15.
    #pragma unroll
    for (int nt = 0; nt < 4; ++nt) {
        const float bvl = bias[nt * 16 + n];
        const int   col = nt * 16 + n;
        if (proj == 2) {
            // Transposed V: consecutive rows contiguous -> packed stores.
            #pragma unroll
            for (int r = 0; r < 4; ++r) {
                v2bf p = pk2bf(acc[nt][2*r] + bvl, acc[nt][2*r + 1] + bvl);
                *(v2bf*)(Vt + ((size_t)b * HD + col) * SEQ + r0 + 8 * hg + 2*r) = p;
            }
        } else {
            __bf16* dst = (proj == 0 ? Qb : Kb) + (size_t)b * SEQ * HD;
            #pragma unroll
            for (int r = 0; r < 8; ++r)
                dst[(size_t)(r0 + r + 8 * hg) * HD + col] = s2bf(acc[nt][r] + bvl);
        }
    }
}

// ---------------------------------------------------------------------------
// Flash attention: one wave per 16-query tile, streaming keys 64 at a time.
// Per chunk: 8 WMMA (S = Q K^T), online softmax (exp2 domain), 8 WMMA (H += P V).
// The 64-wide chunk halves the fixed per-chunk softmax cost (shuffle reductions,
// alpha exps, accumulator rescale) per key versus a 32-wide chunk.
// ---------------------------------------------------------------------------
__global__ void __launch_bounds__(128)
xattn_flash_kernel(const __bf16* __restrict__ Qb, const __bf16* __restrict__ Kb,
                   const __bf16* __restrict__ Vt, float* __restrict__ out) {
    __shared__ __bf16 lds_p[4][16 * 64];     // per-wave P staging (C->A transpose)

    const int lane = threadIdx.x & 31;
    const int wid  = threadIdx.x >> 5;
    const int w    = blockIdx.x * 4 + wid;   // 0..1023
    const int b    = w >> 6;
    const int q0   = (w & 63) << 4;

    const __bf16* Q = Qb + (size_t)b * SEQ * HD;
    const __bf16* K = Kb + (size_t)b * SEQ * HD;
    const __bf16* V = Vt + (size_t)b * HD * SEQ;

    const int n     = lane & 15;
    const int hg    = lane >> 4;
    const int abase = hg ? 16 : 0;
    __bf16* myp = lds_p[wid];

    // logits scale folded with log2(e): softmax computed in exp2 domain.
    const float SCL = 0.125f * 1.44269504088896341f;

    // Q A-fragments for both K=32 contraction chunks (kept in regs)
    v16bf aq[2];
    #pragma unroll
    for (int kc = 0; kc < 2; ++kc) {
        const __bf16* qrow = Q + (size_t)(q0 + n) * HD + kc * 32;
        v8bf lo = *(const v8bf*)(qrow + abase);
        v8bf hi = *(const v8bf*)(qrow + abase + 8);
        #pragma unroll
        for (int i = 0; i < 8; ++i) { aq[kc][i] = lo[i]; aq[kc][i + 8] = hi[i]; }
    }

    float mrun[8], lrun[8];
    v8f   h[4];
    #pragma unroll
    for (int r = 0; r < 8; ++r) { mrun[r] = -__builtin_inff(); lrun[r] = 0.f; }
    #pragma unroll
    for (int nt = 0; nt < 4; ++nt) h[nt] = v8f{0,0,0,0,0,0,0,0};

    for (int j0 = 0; j0 < SEQ; j0 += 64) {
        // ---- S = Q @ K^T for 64 keys (four 16-key N tiles) ----
        v8f s[4];
        #pragma unroll
        for (int nt = 0; nt < 4; ++nt) {
            const __bf16* kbase = K + (size_t)(j0 + nt * 16 + n) * HD + hg * 16;
            v16bf bk0 = *(const v16bf*)(kbase);        // contraction dims 0..31
            v16bf bk1 = *(const v16bf*)(kbase + 32);   // contraction dims 32..63
            s[nt] = __builtin_amdgcn_wmma_f32_16x16x32_bf16(
                false, aq[0], false, bk0, (short)0, v8f{0,0,0,0,0,0,0,0}, false, false);
            s[nt] = __builtin_amdgcn_wmma_f32_16x16x32_bf16(
                false, aq[1], false, bk1, (short)0, s[nt], false, false);
        }

        // ---- online softmax (rows r+8*hg live in 16-lane half-groups) ----
        float cand[8];
        #pragma unroll
        for (int r = 0; r < 8; ++r) {
            #pragma unroll
            for (int nt = 0; nt < 4; ++nt) s[nt][r] *= SCL;
            cand[r] = fmaxf(fmaxf(s[0][r], s[1][r]), fmaxf(s[2][r], s[3][r]));
        }
        #pragma unroll
        for (int r = 0; r < 8; ++r) {
            #pragma unroll
            for (int off = 1; off < 16; off <<= 1)
                cand[r] = fmaxf(cand[r], __shfl_xor(cand[r], off, 32));
        }

        float rowsum[8];
        #pragma unroll
        for (int r = 0; r < 8; ++r) {
            const float mn    = fmaxf(mrun[r], cand[r]);
            const float alpha = exp2f(mrun[r] - mn);    // exp2(-inf)=0 first pass
            mrun[r] = mn;
            lrun[r] *= alpha;
            #pragma unroll
            for (int nt = 0; nt < 4; ++nt) h[nt][r] *= alpha;
            const int row = r + 8 * hg;
            float acc = 0.f;
            #pragma unroll
            for (int nt = 0; nt < 4; ++nt) {
                const float p = exp2f(s[nt][r] - mn);
                acc += p;
                myp[row * 64 + nt * 16 + n] = s2bf(p);
            }
            rowsum[r] = acc;
        }
        #pragma unroll
        for (int r = 0; r < 8; ++r) {
            #pragma unroll
            for (int off = 1; off < 16; off <<= 1)
                rowsum[r] += __shfl_xor(rowsum[r], off, 32);
            lrun[r] += rowsum[r];
        }

        // ---- transpose P (C layout -> A layout) through LDS ----
        asm volatile("s_wait_dscnt 0" ::: "memory");
        v16bf ap[2];
        #pragma unroll
        for (int kc = 0; kc < 2; ++kc) {
            const __bf16* prow = myp + n * 64 + kc * 32;
            v8bf lo = *(const v8bf*)(prow + abase);
            v8bf hi = *(const v8bf*)(prow + abase + 8);
            #pragma unroll
            for (int i = 0; i < 8; ++i) { ap[kc][i] = lo[i]; ap[kc][i + 8] = hi[i]; }
        }

        // ---- H += P @ V (V stored transposed: contiguous B-frag loads) ----
        #pragma unroll
        for (int nt = 0; nt < 4; ++nt) {
            const __bf16* vbase = V + (size_t)(nt * 16 + n) * SEQ + j0 + hg * 16;
            v16bf bv0 = *(const v16bf*)(vbase);        // keys j0+0..31
            v16bf bv1 = *(const v16bf*)(vbase + 32);   // keys j0+32..63
            h[nt] = __builtin_amdgcn_wmma_f32_16x16x32_bf16(
                false, ap[0], false, bv0, (short)0, h[nt], false, false);
            h[nt] = __builtin_amdgcn_wmma_f32_16x16x32_bf16(
                false, ap[1], false, bv1, (short)0, h[nt], false, false);
        }
    }

    // ---- normalize and store H (f32) ----
    float inv[8];
    #pragma unroll
    for (int r = 0; r < 8; ++r) inv[r] = 1.0f / lrun[r];
    #pragma unroll
    for (int nt = 0; nt < 4; ++nt) {
        #pragma unroll
        for (int r = 0; r < 8; ++r) {
            const int row = q0 + r + 8 * hg;
            out[((size_t)b * SEQ + row) * HD + nt * 16 + n] = h[nt][r] * inv[r];
        }
    }
}

// ---------------------------------------------------------------------------
extern "C" void kernel_launch(void* const* d_in, const int* in_sizes, int n_in,
                              void* d_out, int out_size, void* d_ws, size_t ws_size,
                              hipStream_t stream) {
    const float* dec = (const float*)d_in[0];
    const float* enc = (const float*)d_in[1];
    const float* Wq  = (const float*)d_in[2];
    const float* bq  = (const float*)d_in[3];
    const float* Wk  = (const float*)d_in[4];
    const float* bk  = (const float*)d_in[5];
    const float* Wv  = (const float*)d_in[6];
    const float* bv  = (const float*)d_in[7];

    const size_t w_elems  = (size_t)3 * HD * EMBED;  // 147456 bf16 (288KB, 32B-mult)
    const size_t qk_elems = (size_t)NB * SEQ * HD;   // 1M bf16 = 2MB each
    __bf16* Wb = (__bf16*)d_ws;
    __bf16* Qb = Wb + w_elems;
    __bf16* Kb = Qb + qk_elems;
    __bf16* Vt = Kb + qk_elems;                      // ~6.3MB of workspace total

    // Convert weights to bf16 once (73728 pairs).
    xattn_prep_w<<<dim3(288), 256, 0, stream>>>(Wq, Wk, Wv, Wb);

    // 1024 row tiles, 4 waves/block -> 256 blocks; grid.y selects Q/K/V.
    xattn_proj_kernel<<<dim3(256, 3), 128, 0, stream>>>(
        dec, enc, Wb, bq, bk, bv, Qb, Kb, Vt);

    // 1024 query tiles, 4 waves/block -> 256 blocks.
    xattn_flash_kernel<<<dim3(256), 128, 0, stream>>>(Qb, Kb, Vt, (float*)d_out);
}
